// SimpleGAT_38637525795509
// MI455X (gfx1250) — matrix operand order repaired and checked by reading, not compiled, not run on hardware
//
#include <hip/hip_runtime.h>

// ---------------------------------------------------------------------------
// 2-layer GAT for MI455X (gfx1250, wave32).
// GEMMs: v_wmma_f32_16x16x32_bf16, weights pre-packed into B-fragment order
// so every lane loads its fragment with contiguous b128 loads. Branch-free
// inner loop (tail tiles clamped -> redundant identical writes, no EXEC ops).
// Attention logits use (x@Wd)@a_d == x@(Wd@a_d) -> no hd GEMM at all.
// Edge softmax: 3 passes (atomicMax on monotonic uint key, exp+atomicAdd,
// normalize + per-feature global_atomic_add_f32 scatter, one wave per edge).
// ---------------------------------------------------------------------------

typedef __attribute__((ext_vector_type(16))) __bf16        v16bf;
typedef __attribute__((ext_vector_type(8)))  float         v8f;
typedef __attribute__((ext_vector_type(8)))  unsigned int  v8u;

__device__ __forceinline__ unsigned short f2bf(float f) {
    unsigned int u = __float_as_uint(f);
    u += 0x7FFFu + ((u >> 16) & 1u);          // round-to-nearest-even
    return (unsigned short)(u >> 16);
}

// monotonic float->uint key so unsigned atomicMax == float max
__device__ __forceinline__ unsigned int fenc(float f) {
    unsigned int u = __float_as_uint(f);
    return (u & 0x80000000u) ? ~u : (u | 0x80000000u);
}
__device__ __forceinline__ float fdec(unsigned int k) {
    return __uint_as_float((k & 0x80000000u) ? (k & 0x7FFFFFFFu) : ~k);
}

// ---------------------------------------------------------------------------
// float -> bf16 conversion (for activation matrices)
__global__ void __launch_bounds__(256) to_bf16_kernel(
        const float* __restrict__ in, unsigned short* __restrict__ out, size_t n) {
    size_t i = (size_t)blockIdx.x * blockDim.x + threadIdx.x;
    if (i < n) out[i] = f2bf(in[i]);
}

// Pack weight W (K x N, f32 row-major) into WMMA B-fragment order (bf16):
//   dword index = ((ntile*KS + kbi)*32 + lane)*8 + v
//   lane: l15 = col-in-tile, hi = lane>>4; dword v holds K = kbi*32+hi*16+2v (+1)
// -> in the GEMM, each lane's 8-dword fragment is fully contiguous (32 B).
__global__ void __launch_bounds__(256) pack_b_kernel(
        const float* __restrict__ W, unsigned int* __restrict__ out, int K, int N) {
    const int KS = K >> 5;                       // k-steps of 32
    const size_t total = (size_t)(N >> 4) * KS * 32 * 8;
    size_t idx = (size_t)blockIdx.x * blockDim.x + threadIdx.x;
    if (idx >= total) return;
    const int v    = (int)(idx & 7);
    const int lane = (int)((idx >> 3) & 31);
    const int t    = (int)(idx >> 8);
    const int kbi  = t % KS;
    const int nt   = t / KS;
    const int l15  = lane & 15;
    const int hi   = lane >> 4;
    const int krow = kbi * 32 + hi * 16 + 2 * v;
    const int col  = nt * 16 + l15;
    unsigned int lo   = f2bf(W[(size_t)krow * N + col]);
    unsigned int hi16 = f2bf(W[(size_t)(krow + 1) * N + col]);
    out[idx] = lo | (hi16 << 16);
}

// out[k] = sum_c W[k,c] * a[c]      (W: K x C row-major)
__global__ void __launch_bounds__(128) matvec_kernel(
        const float* __restrict__ W, const float* __restrict__ a,
        float* __restrict__ out, int K, int C) {
    int k = blockIdx.x * blockDim.x + threadIdx.x;
    if (k >= K) return;
    float s = 0.f;
    for (int c = 0; c < C; ++c) s += W[(size_t)k * C + c] * a[c];
    out[k] = s;
}

// es[n] = dot(X[n,:], wsv), ed[n] = dot(X[n,:], wdv); one wave per node
__global__ void __launch_bounds__(256) node_logits_kernel(
        const float* __restrict__ X, const float* __restrict__ wsv,
        const float* __restrict__ wdv, float* __restrict__ es,
        float* __restrict__ ed, int N, int C) {
    int w    = blockIdx.x * (blockDim.x >> 5) + (threadIdx.x >> 5);
    int lane = threadIdx.x & 31;
    if (w >= N) return;
    const float* xr = X + (size_t)w * C;
    float s = 0.f, d = 0.f;
    for (int f = lane; f < C; f += 32) {
        float xv = xr[f];
        s += xv * wsv[f];
        d += xv * wdv[f];
    }
    #pragma unroll
    for (int off = 16; off > 0; off >>= 1) {
        s += __shfl_xor(s, off, 32);
        d += __shfl_xor(d, off, 32);
    }
    if (lane == 0) { es[w] = s; ed[w] = d; }
}

// ---------------------------------------------------------------------------
// C[M,N] = A[M,K] @ B[K,N]; A bf16 row-major, B pre-packed fragments.
// Requires M % 16 == 0 (true here: 50000 = 3125*16). Each wave: 4 M-tiles
// (64 rows) x one 16-col strip, B fragment registers reused across the 4
// WMMAs per k-step. Tail waves clamp to the last tile (identical redundant
// writes) so there are NO branches and EXEC is never touched around WMMA.
template <int KT>
__global__ void __launch_bounds__(256) gemm_bf16_wmma(
        const unsigned short* __restrict__ A,   // M x K bf16
        const unsigned int*  __restrict__ Bp,   // packed B fragments
        float* __restrict__ C,                  // M x N f32
        int M, int N, int Krt) {
    constexpr int MREP = 4;
    const int K     = KT ? KT : Krt;
    const int KS    = K >> 5;
    const int MT    = M >> 4;                        // full 16-row tiles
    const int lane  = threadIdx.x & 31;
    const int l15   = lane & 15;
    const int hi    = lane >> 4;                     // 0 or 1 (half-wave)
    const int ntile = blockIdx.y;
    const int mtile0 = (blockIdx.x * 8 + (threadIdx.x >> 5)) * MREP;

    int mt[MREP];
    #pragma unroll
    for (int r = 0; r < MREP; ++r) {
        int t = mtile0 + r;
        mt[r] = (t < MT) ? t : (MT - 1);             // clamp: no branches later
    }

    const int col = ntile * 16 + l15;
    const unsigned int* bbase = Bp + ((size_t)ntile * KS * 32 + lane) * 8;

    v8f acc[MREP] = {};
    #pragma unroll
    for (int kbi = 0; kbi < (KT ? KT / 32 : 64); ++kbi) {
        if (!KT && kbi >= KS) break;
        // B fragment: one contiguous 32-byte load per lane (b128 x2)
        const v8u bu = *(const v8u*)(bbase + (size_t)kbi * 256);
        const v16bf bv = __builtin_bit_cast(v16bf, bu);
        #pragma unroll
        for (int r = 0; r < MREP; ++r) {
            const int arow = mt[r] * 16 + l15;
            // A row segment: 4 chunks of 16 B; half 'hi' needs chunks 2hi, 2hi+1
            const uint4* ap = (const uint4*)(A + (size_t)arow * K + kbi * 32);
            uint4 a0 = ap[2 * hi];
            uint4 a1 = ap[2 * hi + 1];
            v8u au;
            au[0] = a0.x; au[1] = a0.y; au[2] = a0.z; au[3] = a0.w;
            au[4] = a1.x; au[5] = a1.y; au[6] = a1.z; au[7] = a1.w;
            acc[r] = __builtin_amdgcn_wmma_f32_16x16x32_bf16(
                         false, __builtin_bit_cast(v16bf, au),
                         false, bv, (short)0, acc[r], false, false);
        }
    }

    // C/D layout: VGPR q -> row q (lanes 0-15) / q+8 (lanes 16-31), col = l15
    #pragma unroll
    for (int r = 0; r < MREP; ++r) {
        const int mbase = mt[r] * 16 + hi * 8;
        float* cp = C + (size_t)mbase * N + col;
        #pragma unroll
        for (int q = 0; q < 8; ++q)
            cp[(size_t)q * N] = acc[r][q];
    }
}

// ---------------------------------------------------------------------------
// Edge pass 1: e = leaky_relu(es[src]+ed[dst], 0.2); segment-max via atomicMax
__global__ void __launch_bounds__(256) edge_logits_kernel(
        const int* __restrict__ src, const int* __restrict__ dst,
        const float* __restrict__ es, const float* __restrict__ ed,
        float* __restrict__ ebuf, unsigned int* __restrict__ menc, int E) {
    int e = blockIdx.x * blockDim.x + threadIdx.x;
    if (e >= E) return;
    int s = src[e], d = dst[e];
    float v = es[s] + ed[d];
    v = (v > 0.f) ? v : 0.2f * v;
    ebuf[e] = v;
    atomicMax(&menc[d], fenc(v));
}

// Edge pass 2: ex = exp(e - max[dst]); denom[dst] += ex
__global__ void __launch_bounds__(256) edge_exp_kernel(
        const int* __restrict__ dst, const float* __restrict__ ebuf,
        const unsigned int* __restrict__ menc, float* __restrict__ exbuf,
        float* __restrict__ denom, int E) {
    int e = blockIdx.x * blockDim.x + threadIdx.x;
    if (e >= E) return;
    int d = dst[e];
    float ex = expf(ebuf[e] - fdec(menc[d]));
    exbuf[e] = ex;
    atomicAdd(&denom[d], ex);
}

// Edge pass 3: agg[dst,:] += (ex/denom[dst]) * hs[src,:]; one wave per edge
__global__ void __launch_bounds__(256) edge_scatter_kernel(
        const int* __restrict__ src, const int* __restrict__ dst,
        const float* __restrict__ exbuf, const float* __restrict__ denom,
        const float* __restrict__ hs, float* __restrict__ agg, int E, int C) {
    int e    = blockIdx.x * (blockDim.x >> 5) + (threadIdx.x >> 5);
    int lane = threadIdx.x & 31;
    if (e >= E) return;
    int s = src[e], d = dst[e];
    float alpha = exbuf[e] / denom[d];
    const float* hr = hs  + (size_t)s * C;
    float*       ar = agg + (size_t)d * C;
    for (int f = lane; f < C; f += 32)
        atomicAdd(&ar[f], alpha * hr[f]);
}

// Layer-1 epilogue: h = relu(agg + b + skip + lb); also emit bf16 copy
__global__ void __launch_bounds__(256) epilogue1_kernel(
        const float* __restrict__ agg, const float* __restrict__ skip,
        const float* __restrict__ b, const float* __restrict__ lb,
        float* __restrict__ h, unsigned short* __restrict__ hb,
        int N, int C) {
    size_t i = (size_t)blockIdx.x * blockDim.x + threadIdx.x;
    if (i >= (size_t)N * C) return;
    int c = (int)(i % C);
    float v = agg[i] + b[c] + skip[i] + lb[c];
    v = fmaxf(v, 0.f);
    h[i]  = v;
    hb[i] = f2bf(v);
}

// Layer-2 epilogue: out = agg + b + skip + lb (no relu)
__global__ void __launch_bounds__(256) epilogue2_kernel(
        const float* __restrict__ agg, const float* __restrict__ skip,
        const float* __restrict__ b, const float* __restrict__ lb,
        float* __restrict__ out, int N, int C) {
    size_t i = (size_t)blockIdx.x * blockDim.x + threadIdx.x;
    if (i >= (size_t)N * C) return;
    int c = (int)(i % C);
    out[i] = agg[i] + b[c] + skip[i] + lb[c];
}

// ---------------------------------------------------------------------------
extern "C" void kernel_launch(void* const* d_in, const int* in_sizes, int n_in,
                              void* d_out, int out_size, void* d_ws, size_t ws_size,
                              hipStream_t stream) {
    const float* x   = (const float*)d_in[0];
    const int*   ei  = (const int*)  d_in[1];
    const float* W1s = (const float*)d_in[2];
    const float* W1d = (const float*)d_in[3];
    const float* a1s = (const float*)d_in[4];
    const float* a1d = (const float*)d_in[5];
    const float* b1  = (const float*)d_in[6];
    const float* L1w = (const float*)d_in[7];
    const float* L1b = (const float*)d_in[8];
    const float* W2s = (const float*)d_in[9];
    const float* W2d = (const float*)d_in[10];
    const float* a2s = (const float*)d_in[11];
    const float* a2d = (const float*)d_in[12];
    const float* b2  = (const float*)d_in[13];
    const float* L2w = (const float*)d_in[14];
    const float* L2b = (const float*)d_in[15];

    const int DH  = in_sizes[4];                 // 128
    const int DO  = in_sizes[11];                // 64
    const int DIN = in_sizes[2] / DH;            // 128
    const int N   = in_sizes[0] / DIN;           // 50000
    const int E   = in_sizes[1] / 2;             // 800000
    const int* src = ei;
    const int* dst = ei + E;

    // ---- workspace layout ----
    char* ws = (char*)d_ws;
    size_t off = 0;
    auto alloc = [&](size_t bytes) -> void* {
        void* p = ws + off;
        off = (off + bytes + 255) & ~(size_t)255;
        return p;
    };
    unsigned short* xb  = (unsigned short*)alloc((size_t)N * DIN * 2); // also hb
    unsigned int* w1sp  = (unsigned int*)alloc((size_t)DIN * DH * 2);  // packed
    unsigned int* l1wp  = (unsigned int*)alloc((size_t)DIN * DH * 2);
    unsigned int* w2sp  = (unsigned int*)alloc((size_t)DH * DO * 2);
    unsigned int* l2wp  = (unsigned int*)alloc((size_t)DH * DO * 2);
    float* wsv1  = (float*)alloc((size_t)DIN * 4);
    float* wdv1  = (float*)alloc((size_t)DIN * 4);
    float* wsv2  = (float*)alloc((size_t)DH * 4);
    float* wdv2  = (float*)alloc((size_t)DH * 4);
    float* hs    = (float*)alloc((size_t)N * DH * 4);   // layer2 reuses (N*DO)
    float* skip  = (float*)alloc((size_t)N * DH * 4);
    float* agg   = (float*)alloc((size_t)N * DH * 4);
    float* h     = (float*)alloc((size_t)N * DH * 4);
    float* es    = (float*)alloc((size_t)N * 4);
    float* ed    = (float*)alloc((size_t)N * 4);
    unsigned int* menc = (unsigned int*)alloc((size_t)N * 4);
    float* denom = (float*)alloc((size_t)N * 4);
    float* ebuf  = (float*)alloc((size_t)E * 4);
    float* exbuf = (float*)alloc((size_t)E * 4);
    unsigned short* hb = xb;   // reuse: xb dead after layer-1 GEMMs

    const int TB = 256;
    auto cdiv = [](long long a, long long b) { return (unsigned)((a + b - 1) / b); };

    // ---- precompute: bf16 convert + weight fragment packing + matvecs ----
    to_bf16_kernel<<<cdiv((size_t)N * DIN, TB), TB, 0, stream>>>(x, xb, (size_t)N * DIN);
    pack_b_kernel<<<cdiv((size_t)DIN * DH / 2, TB), TB, 0, stream>>>(W1s, w1sp, DIN, DH);
    pack_b_kernel<<<cdiv((size_t)DIN * DH / 2, TB), TB, 0, stream>>>(L1w, l1wp, DIN, DH);
    pack_b_kernel<<<cdiv((size_t)DH * DO / 2, TB), TB, 0, stream>>>(W2s, w2sp, DH, DO);
    pack_b_kernel<<<cdiv((size_t)DH * DO / 2, TB), TB, 0, stream>>>(L2w, l2wp, DH, DO);
    matvec_kernel<<<cdiv(DIN, 128), 128, 0, stream>>>(W1s, a1s, wsv1, DIN, DH);
    matvec_kernel<<<cdiv(DIN, 128), 128, 0, stream>>>(W1d, a1d, wdv1, DIN, DH);
    matvec_kernel<<<cdiv(DH, 128), 128, 0, stream>>>(W2s, a2s, wsv2, DH, DO);
    matvec_kernel<<<cdiv(DH, 128), 128, 0, stream>>>(W2d, a2d, wdv2, DH, DO);

    const unsigned MT = cdiv(N, 16);
    dim3 g1(cdiv(MT, 32), DH / 16);        // 8 waves/block x 4 M-tiles/wave
    dim3 g2(cdiv(MT, 32), DO / 16);
    const unsigned gE  = cdiv(E, TB);
    const unsigned gEw = cdiv(E, 8);       // one wave per edge, 8 waves/block
    const unsigned gNw = cdiv(N, 8);

    // ======================== layer 1 ========================
    hipMemsetAsync(menc,  0, (size_t)N * 4, stream);
    hipMemsetAsync(denom, 0, (size_t)N * 4, stream);
    hipMemsetAsync(agg,   0, (size_t)N * DH * 4, stream);

    if (DIN == 128) {
        gemm_bf16_wmma<128><<<g1, TB, 0, stream>>>(xb, w1sp, hs,   N, DH, DIN);
        gemm_bf16_wmma<128><<<g1, TB, 0, stream>>>(xb, l1wp, skip, N, DH, DIN);
    } else {
        gemm_bf16_wmma<0><<<g1, TB, 0, stream>>>(xb, w1sp, hs,   N, DH, DIN);
        gemm_bf16_wmma<0><<<g1, TB, 0, stream>>>(xb, l1wp, skip, N, DH, DIN);
    }
    node_logits_kernel<<<gNw, TB, 0, stream>>>(x, wsv1, wdv1, es, ed, N, DIN);
    edge_logits_kernel<<<gE, TB, 0, stream>>>(src, dst, es, ed, ebuf, menc, E);
    edge_exp_kernel<<<gE, TB, 0, stream>>>(dst, ebuf, menc, exbuf, denom, E);
    edge_scatter_kernel<<<gEw, TB, 0, stream>>>(src, dst, exbuf, denom, hs, agg, E, DH);
    epilogue1_kernel<<<cdiv((size_t)N * DH, TB), TB, 0, stream>>>(agg, skip, b1, L1b, h, hb, N, DH);

    // ======================== layer 2 ========================
    hipMemsetAsync(menc,  0, (size_t)N * 4, stream);
    hipMemsetAsync(denom, 0, (size_t)N * 4, stream);
    hipMemsetAsync(agg,   0, (size_t)N * DO * 4, stream);

    if (DH == 128) {
        gemm_bf16_wmma<128><<<g2, TB, 0, stream>>>(hb, w2sp, hs,   N, DO, DH);
        gemm_bf16_wmma<128><<<g2, TB, 0, stream>>>(hb, l2wp, skip, N, DO, DH);
    } else {
        gemm_bf16_wmma<0><<<g2, TB, 0, stream>>>(hb, w2sp, hs,   N, DO, DH);
        gemm_bf16_wmma<0><<<g2, TB, 0, stream>>>(hb, l2wp, skip, N, DO, DH);
    }
    node_logits_kernel<<<gNw, TB, 0, stream>>>(h, wsv2, wdv2, es, ed, N, DH);
    edge_logits_kernel<<<gE, TB, 0, stream>>>(src, dst, es, ed, ebuf, menc, E);
    edge_exp_kernel<<<gE, TB, 0, stream>>>(dst, ebuf, menc, exbuf, denom, E);
    edge_scatter_kernel<<<gEw, TB, 0, stream>>>(src, dst, exbuf, denom, hs, agg, E, DO);
    epilogue2_kernel<<<cdiv((size_t)N * DO, TB), TB, 0, stream>>>(agg, skip, b2, L2b, (float*)d_out, N, DO);
}
